// NIVR_17738214932547
// MI455X (gfx1250) — matrix operand order repaired and checked by reading, not compiled, not run on hardware
//
#include <hip/hip_runtime.h>
#include <hip/hip_bf16.h>

typedef __attribute__((ext_vector_type(16))) _Float16 v16h;
typedef __attribute__((ext_vector_type(8)))  float    v8f;
typedef __attribute__((ext_vector_type(4)))  float    v4f;

#define SIDE      512
#define N_PIX     (SIDE * SIDE)
#define TIME_LEN  300
#define ROWS      64          // pixels per block (4 m-tiles)
#define THREADS   128         // 4 waves of 32
#define K1        192         // 168 padded to 6*32
#define K1T       6
#define K2T       16
#define K3T       16
#define PI_F      3.14159265358979323846f

// ---------------- workspace layout (bytes) ----------------
#define PHI_OFF   0u                          // 128 f16
#define W1H_OFF   1024u                       // 6*32 tiles * 1KB = 196608
#define W2H_OFF   (1024u + 196608u)           // 16*32 tiles * 1KB = 524288
#define W3H_OFF   (1024u + 196608u + 524288u) // 16 tiles * 1KB = 16384

// ---------------- time-branch MLP: phi_t[128] in f16 ----------------
__global__ void phi_kernel(const float* __restrict__ W1p, const float* __restrict__ b1p,
                           const float* __restrict__ W2p, const float* __restrict__ b2p,
                           const int* __restrict__ idx, _Float16* __restrict__ phi_out) {
    __shared__ float sR[32];
    __shared__ float sH[256];
    const int tid = threadIdx.x;
    const float t = (float)(*idx) / (float)TIME_LEN;
    if (tid < 32) {
        const int l = tid & 15;
        const float ang = t * (float)(1 << l) * PI_F;
        sR[tid] = (tid < 16) ? __sinf(ang) : __cosf(ang);
    }
    __syncthreads();
    float acc = b1p[tid];
    #pragma unroll 8
    for (int i = 0; i < 32; ++i) acc += sR[i] * W1p[i * 256 + tid];
    sH[tid] = acc > 0.0f ? acc : 0.0f;
    __syncthreads();
    if (tid < 128) {
        float acc2 = b2p[tid];
        #pragma unroll 8
        for (int i = 0; i < 256; ++i) acc2 += sH[i] * W2p[i * 128 + tid];
        phi_out[tid] = (_Float16)acc2;
    }
}

// ---------------- weight f32 -> f16, pre-swizzled to B-fragment layout ----------------
// Tile = 32(K) x 16(N), 512 f16 = 1KB. Within a tile: lane = halfword/16 (0..31),
// j = halfword%16. lane<16 holds K {0..7,16..23} of column n=lane%16; lanes 16..31
// hold K {8..15,24..31}. Tile order: (nt * KT + kt) so the k loop streams contiguously.
__global__ void swizzleB(const float* __restrict__ W, _Float16* __restrict__ out,
                         int Ksrc, int Nsrc, int KT, int NT) {
    const int total = KT * NT * 512;
    for (int e = blockIdx.x * blockDim.x + threadIdx.x; e < total;
         e += gridDim.x * blockDim.x) {
        const int tile = e >> 9;
        const int t    = e & 511;
        const int lane = t >> 4;
        const int j    = t & 15;
        const int nt   = tile / KT;
        const int kt   = tile - nt * KT;
        const int k    = kt * 32 + (lane >> 4) * 8 + (j & 7) + (j >> 3) * 16;
        const int n    = nt * 16 + (lane & 15);
        const float v  = (k < Ksrc && n < Nsrc) ? W[k * Nsrc + n] : 0.0f;
        out[e] = (_Float16)v;
    }
}

// ---------------- 4m x 4n register-blocked GEMM group ----------------
// One wave computes rows 0..63 (4 m-tiles) x 4 n-tiles [ntBase, ntBase+4).
// Per k-step: 4 A frags (LDS) + 4 B frags (global) -> 16 WMMAs.
template <int KT>
__device__ __forceinline__ void gemm_group(
    const _Float16* __restrict__ sIn, int astride,
    _Float16* __restrict__ sOut,
    const _Float16* __restrict__ w,
    const float* __restrict__ bias,
    int ntBase, int lane, int lane16, int laneHi) {

    v8f acc[4][4] = {};
    #pragma unroll 1
    for (int kt = 0; kt < KT; ++kt) {
        // A frags: 16-bit A 16x32 ISA layout.
        // lane<16: M=lane, K {0..7,16..23}; lane>=16: M=lane-16, K {8..15,24..31}
        v16h a[4];
        #pragma unroll
        for (int mt = 0; mt < 4; ++mt) {
            const _Float16* pA = sIn + (mt * 16 + lane16) * astride + kt * 32 + laneHi * 8;
            ((v4f*)&a[mt])[0] = *(const v4f*)(pA);
            ((v4f*)&a[mt])[1] = *(const v4f*)(pA + 16);
        }
        // B frags: pre-swizzled 1KB tiles, 32B per lane, two b128 loads each
        v16h b[4];
        #pragma unroll
        for (int ntl = 0; ntl < 4; ++ntl) {
            const v4f* pB = (const v4f*)(w + ((ntBase + ntl) * KT + kt) * 512) + lane * 2;
            ((v4f*)&b[ntl])[0] = pB[0];
            ((v4f*)&b[ntl])[1] = pB[1];
        }
        __builtin_prefetch(w + (ntBase * KT + kt + 1) * 512, 0, 1);
        #pragma unroll
        for (int mt = 0; mt < 4; ++mt) {
            #pragma unroll
            for (int ntl = 0; ntl < 4; ++ntl) {
                acc[mt][ntl] = __builtin_amdgcn_wmma_f32_16x16x32_f16(
                    false, a[mt], false, b[ntl], (short)0, acc[mt][ntl], false, false);
            }
        }
    }
    // epilogue: bias + ReLU + f32->f16 into row-major LDS (stride 512)
    #pragma unroll
    for (int ntl = 0; ntl < 4; ++ntl) {
        const float bv = bias[(ntBase + ntl) * 16 + lane16];
        #pragma unroll
        for (int mt = 0; mt < 4; ++mt) {
            #pragma unroll
            for (int r = 0; r < 8; ++r) {
                float v = acc[mt][ntl][r] + bv;
                v = v > 0.0f ? v : 0.0f;
                sOut[(mt * 16 + r + laneHi * 8) * 512 + (ntBase + ntl) * 16 + lane16] =
                    (_Float16)v;
            }
        }
    }
}

// ---------------- fused pixel MLP ----------------
__global__ __launch_bounds__(THREADS) void mlp_kernel(
    const _Float16* __restrict__ w1h, const _Float16* __restrict__ w2h,
    const _Float16* __restrict__ w3h, const _Float16* __restrict__ phi,
    const float* __restrict__ b1f, const float* __restrict__ b2f,
    const float* __restrict__ b3f, float* __restrict__ out) {

    // sA: feat (64x192) then h2 (64x512).  sB: h1 (64x512).  128KB LDS total.
    __shared__ __align__(16) _Float16 sA[ROWS * 512];
    __shared__ __align__(16) _Float16 sB[ROWS * 512];

    const int lane   = threadIdx.x & 31;
    const int wave   = threadIdx.x >> 5;
    const int lane16 = lane & 15;
    const int laneHi = lane >> 4;          // 0 | 1
    const int pix0   = blockIdx.x * ROWS;

    // ---- build feat = [posenc(coords/512, L=10) | phi_t | 0-pad] in f16 ----
    const float inv = 1.0f / (float)SIDE;
    for (int e = threadIdx.x; e < ROWS * K1; e += THREADS) {
        const int row = e / K1;
        const int col = e - row * K1;
        const int p   = pix0 + row;
        float val;
        if (col < 40) {
            const int axis  = col / 20;             // 0: x, 1: y
            const int cidx  = col - axis * 20;      // 0..19
            const int l     = cidx % 10;
            const float crd = axis ? (float)(p & (SIDE - 1)) : (float)(p >> 9);
            const float ang = crd * inv * (float)(1 << l) * PI_F;
            val = (cidx >= 10) ? __cosf(ang) : __sinf(ang);
        } else if (col < 168) {
            val = (float)phi[col - 40];
        } else {
            val = 0.0f;
        }
        sA[row * K1 + col] = (_Float16)val;
    }
    __syncthreads();

    // ---------- GEMM1: [64x192] x [192x512] -> h1 (sB), ReLU ----------
    #pragma unroll 1
    for (int g = 0; g < 2; ++g)
        gemm_group<K1T>(sA, K1, sB, w1h, b1f, g * 16 + wave * 4, lane, lane16, laneHi);
    __syncthreads();

    // ---------- GEMM2: [64x512] x [512x512] -> h2 (sA), ReLU ----------
    #pragma unroll 1
    for (int g = 0; g < 2; ++g)
        gemm_group<K2T>(sB, 512, sA, w2h, b2f, g * 16 + wave * 4, lane, lane16, laneHi);
    __syncthreads();

    // ---------- GEMM3: [64x512] x [512x16(pad of 3)] -> rgb ----------
    {
        const int mBase = wave * 16;
        v8f acc = {};
        #pragma unroll
        for (int kt = 0; kt < K3T; ++kt) {
            const _Float16* pA = sA + (mBase + lane16) * 512 + kt * 32 + laneHi * 8;
            v16h a;
            ((v4f*)&a)[0] = *(const v4f*)(pA);
            ((v4f*)&a)[1] = *(const v4f*)(pA + 16);
            const v4f* pB = (const v4f*)(w3h + kt * 512) + lane * 2;
            v16h b;
            ((v4f*)&b)[0] = pB[0];
            ((v4f*)&b)[1] = pB[1];
            acc = __builtin_amdgcn_wmma_f32_16x16x32_f16(
                false, a, false, b, (short)0, acc, false, false);
        }
        if (lane16 < 3) {
            const float bias = b3f[lane16];
            const int   mB   = pix0 + mBase + laneHi * 8;
            #pragma unroll
            for (int r = 0; r < 8; ++r) {
                out[lane16 * N_PIX + mB + r] = acc[r] + bias;
            }
        }
    }
}

extern "C" void kernel_launch(void* const* d_in, const int* in_sizes, int n_in,
                              void* d_out, int out_size, void* d_ws, size_t ws_size,
                              hipStream_t stream) {
    const float* W1p = (const float*)d_in[1];
    const float* b1p = (const float*)d_in[2];
    const float* W2p = (const float*)d_in[3];
    const float* b2p = (const float*)d_in[4];
    const float* W1f = (const float*)d_in[5];
    const float* b1f = (const float*)d_in[6];
    const float* W2f = (const float*)d_in[7];
    const float* b2f = (const float*)d_in[8];
    const float* W3f = (const float*)d_in[9];
    const float* b3f = (const float*)d_in[10];
    const int*   idx = (const int*)d_in[11];
    float* out = (float*)d_out;

    char* ws = (char*)d_ws;
    _Float16* phi = (_Float16*)(ws + PHI_OFF);
    _Float16* w1h = (_Float16*)(ws + W1H_OFF);
    _Float16* w2h = (_Float16*)(ws + W2H_OFF);
    _Float16* w3h = (_Float16*)(ws + W3H_OFF);

    phi_kernel<<<1, 256, 0, stream>>>(W1p, b1p, W2p, b2p, idx, phi);
    swizzleB<<<(K1T * 32 * 512) / 256, 256, 0, stream>>>(W1f, w1h, 168, 512, K1T, 32);
    swizzleB<<<(K2T * 32 * 512) / 256, 256, 0, stream>>>(W2f, w2h, 512, 512, K2T, 32);
    swizzleB<<<(K3T * 1 * 512) / 256, 256, 0, stream>>>(W3f, w3h, 512, 3, K3T, 1);
    mlp_kernel<<<N_PIX / ROWS, THREADS, 0, stream>>>(w1h, w2h, w3h, phi,
                                                     b1f, b2f, b3f, out);
}
